// TemporalSegmentSelection_52553219833980
// MI455X (gfx1250) — matrix operand order repaired and checked by reading, not compiled, not run on hardware
//
#include <hip/hip_runtime.h>
#include <hip/hip_bf16.h>
#include <math.h>

typedef __attribute__((ext_vector_type(16))) _Float16 v16h;
typedef __attribute__((ext_vector_type(8)))  _Float16 v8h;
typedef __attribute__((ext_vector_type(8)))  float    v8f;

#define SEGS   32
#define TOPK   10
#define NHEADS 4
#define B_     16
#define T_     4096
#define C_     512
#define CLIP   128          // T / SEGS
#define ROWS   (B_ * SEGS)  // 512
#define TWO_C  1024

// ------------------------------------------------------------------
// Kernel 1: per-clip mean of audio & visual, tanh, store fusion f16
// One block per (b, seg); 256 threads; each thread owns 2 channels of
// audio and 2 of visual.  Fully coalesced float2 streaming reads.
// ------------------------------------------------------------------
__global__ void tss_clip_pool_tanh(const float* __restrict__ audio,
                                   const float* __restrict__ visual,
                                   _Float16* __restrict__ fus /* [ROWS, 2C] */) {
    int blk = blockIdx.x;             // b*SEGS + s
    int b = blk >> 5, s = blk & 31;
    int c2 = threadIdx.x * 2;
    const float2* ap = (const float2*)(audio  + (size_t)(b * T_ + s * CLIP) * C_ + c2);
    const float2* vp = (const float2*)(visual + (size_t)(b * T_ + s * CLIP) * C_ + c2);
    float a0 = 0.f, a1 = 0.f, v0 = 0.f, v1 = 0.f;
    for (int r = 0; r < CLIP; ++r) {
        float2 av = ap[(size_t)r * (C_ / 2)];
        float2 vv = vp[(size_t)r * (C_ / 2)];
        a0 += av.x; a1 += av.y; v0 += vv.x; v1 += vv.y;
    }
    const float inv = 1.0f / (float)CLIP;
    _Float16* out = fus + (size_t)blk * TWO_C;
    out[c2]          = (_Float16)tanhf(a0 * inv);
    out[c2 + 1]      = (_Float16)tanhf(a1 * inv);
    out[C_ + c2]     = (_Float16)tanhf(v0 * inv);
    out[C_ + c2 + 1] = (_Float16)tanhf(v1 * inv);
}

// ------------------------------------------------------------------
// Kernel 2: f32 -> f16 conversion (weights / qst)
// ------------------------------------------------------------------
__global__ void tss_f32_to_f16(const float* __restrict__ in,
                               _Float16* __restrict__ out, int n) {
    int i = blockIdx.x * blockDim.x + threadIdx.x;
    if (i < n) out[i] = (_Float16)in[i];
}

// ------------------------------------------------------------------
// Kernel 3: generic  D[M,N] = A[M,K] (f16) x W[N,K]^T (f16) + bias,
// f32 accumulation via v_wmma_f32_16x16x32_f16.
// One wave per 16x16 tile; wave-uniform early exit keeps EXEC all-1s.
// 16-bit A/B fragments use contiguous k = 16*hi + 0..15 in both; C/D per CDNA5 ISA 7.12.2 (
// 32-bit C/D 16x16).
// ------------------------------------------------------------------
__global__ void tss_wmma_gemm_nt(const _Float16* __restrict__ A,  // [M,K]
                                 const _Float16* __restrict__ W,  // [N,K]
                                 const float*    __restrict__ bias, // [N]
                                 float*          __restrict__ Df, // [M,N] or null
                                 _Float16*       __restrict__ Dh, // [M,N] or null
                                 int M, int N, int K) {
    int gwave = (int)((blockIdx.x * blockDim.x + threadIdx.x) >> 5);
    int lane  = threadIdx.x & 31;
    int ntile_n = N >> 4;
    int tiles   = (M >> 4) * ntile_n;
    if (gwave >= tiles) return;                 // wave-uniform
    int tm = (gwave / ntile_n) << 4;
    int tn = (gwave % ntile_n) << 4;
    int r15 = lane & 15;
    int hi  = lane >> 4;                        // 0 for lanes 0-15, 1 for 16-31

    // A: lane holds row (tm+r15); halves 0..15 at K = 16*hi + 0..15
    //    -> one contiguous 32B load.
    const _Float16* arow = A + (size_t)(tm + r15) * K + 16 * hi;
    // B: lane holds column (tn+r15); halves 0..15 at K = 16*hi + 0..15
    //    -> one contiguous 32B load.
    const _Float16* brow = W + (size_t)(tn + r15) * K + 16 * hi;

    v8f acc = {};
    for (int k0 = 0; k0 < K; k0 += 32) {
        v16h a = *(const v16h*)(arow + k0);
        v16h b = *(const v16h*)(brow + k0);
        acc = __builtin_amdgcn_wmma_f32_16x16x32_f16(
                  false, a, false, b, (short)0, acc, false, false);
    }

    // D: lane -> column n = r15; VGPR v -> row m = v + 8*hi
    int n = tn + r15;
    float bv = bias ? bias[n] : 0.0f;
#pragma unroll
    for (int v = 0; v < 8; ++v) {
        int m = tm + v + 8 * hi;
        float val = acc[v] + bv;
        if (Df) Df[(size_t)m * N + n] = val;
        if (Dh) Dh[(size_t)m * N + n] = (_Float16)val;
    }
}

// ------------------------------------------------------------------
// Kernel 4: attention scores -> softmax -> head mean -> top-10 sorted.
// One block per batch, 128 threads = (head, seg) pairs.
// ------------------------------------------------------------------
__global__ void tss_attn_topk(const float* __restrict__ q,   // [B, C]
                              const float* __restrict__ k,   // [B*SEGS, C]
                              int*   __restrict__ idx_ws,    // [B, TOPK]
                              float* __restrict__ idx_out) { // [B, TOPK]
    __shared__ float sc[NHEADS][SEGS];
    __shared__ float tw[SEGS];
    int b   = blockIdx.x;
    int tid = threadIdx.x;          // 0..127
    int h = tid >> 5, s = tid & 31;
    const int dh = C_ / NHEADS;     // 128

    const float* qp = q + (size_t)b * C_ + h * dh;
    const float* kp = k + ((size_t)b * SEGS + s) * C_ + h * dh;
    float d = 0.f;
    for (int i = 0; i < dh; ++i) d += qp[i] * kp[i];
    sc[h][s] = d * 0.08838834764831845f;   // 1/sqrt(128)
    __syncthreads();

    float mx = -1e30f;
    for (int j = 0; j < SEGS; ++j) mx = fmaxf(mx, sc[h][j]);
    __syncthreads();
    float e = __expf(sc[h][s] - mx);
    sc[h][s] = e;
    __syncthreads();
    float sum = 0.f;
    for (int j = 0; j < SEGS; ++j) sum += sc[h][j];
    float p = e / sum;
    __syncthreads();
    sc[h][s] = p;
    __syncthreads();
    if (tid < SEGS)
        tw[tid] = 0.25f * (sc[0][tid] + sc[1][tid] + sc[2][tid] + sc[3][tid]);
    __syncthreads();

    if (tid == 0) {
        int sel[TOPK];
        unsigned used = 0;
        for (int t = 0; t < TOPK; ++t) {        // strict '>' keeps lowest index on ties
            float best = -1e30f; int bi = 0;
            for (int j = 0; j < SEGS; ++j)
                if (!((used >> j) & 1u) && tw[j] > best) { best = tw[j]; bi = j; }
            used |= 1u << bi; sel[t] = bi;
        }
        for (int i = 1; i < TOPK; ++i) {        // sort ascending
            int v = sel[i], j = i - 1;
            while (j >= 0 && sel[j] > v) { sel[j + 1] = sel[j]; --j; }
            sel[j + 1] = v;
        }
        for (int t = 0; t < TOPK; ++t) {
            idx_ws[b * TOPK + t]  = sel[t];
            idx_out[b * TOPK + t] = (float)sel[t];
        }
    }
}

// ------------------------------------------------------------------
// Kernel 5: gather selected segments (float4 row copies, 168 MB total).
// One block per output row; threads 0-127 copy audio, 128-255 visual.
// ------------------------------------------------------------------
__global__ void tss_gather_rows(const float* __restrict__ audio,
                                const float* __restrict__ visual,
                                const int* __restrict__ idx_ws,
                                float* __restrict__ outA,
                                float* __restrict__ outV) {
    int blk = blockIdx.x;                 // b*(TOPK*CLIP) + i
    int b = blk / (TOPK * CLIP);
    int i = blk % (TOPK * CLIP);
    int seg = idx_ws[b * TOPK + i / CLIP];
    int t = seg * CLIP + (i & (CLIP - 1));
    const float4* srcA = (const float4*)(audio  + ((size_t)b * T_ + t) * C_);
    const float4* srcV = (const float4*)(visual + ((size_t)b * T_ + t) * C_);
    float4* dA = (float4*)(outA + ((size_t)b * (TOPK * CLIP) + i) * C_);
    float4* dV = (float4*)(outV + ((size_t)b * (TOPK * CLIP) + i) * C_);
    int tid = threadIdx.x;
    if (tid < 128) dA[tid] = srcA[tid];
    else           dV[tid - 128] = srcV[tid - 128];
}

// ------------------------------------------------------------------
extern "C" void kernel_launch(void* const* d_in, const int* in_sizes, int n_in,
                              void* d_out, int out_size, void* d_ws, size_t ws_size,
                              hipStream_t stream) {
    (void)in_sizes; (void)n_in; (void)out_size; (void)ws_size;
    const float* audio  = (const float*)d_in[0];
    const float* visual = (const float*)d_in[1];
    const float* qst    = (const float*)d_in[2];
    const float* fc_w   = (const float*)d_in[3];
    const float* fc_b   = (const float*)d_in[4];
    const float* wq     = (const float*)d_in[5];
    const float* bq     = (const float*)d_in[6];
    const float* wk     = (const float*)d_in[7];
    const float* bk     = (const float*)d_in[8];

    char* ws = (char*)d_ws;
    _Float16* fus_h = (_Float16*)(ws + 0x000000); // [512,1024] 1 MB
    _Float16* fcw_h = (_Float16*)(ws + 0x100000); // [512,1024] 1 MB
    _Float16* wq_h  = (_Float16*)(ws + 0x200000); // [512,512] 512 KB
    _Float16* wk_h  = (_Float16*)(ws + 0x280000); // [512,512] 512 KB
    _Float16* qst_h = (_Float16*)(ws + 0x300000); // [16,512]  16 KB
    _Float16* kv_h  = (_Float16*)(ws + 0x310000); // [512,512] 512 KB
    float*    q_f   = (float*)   (ws + 0x390000); // [16,512]  32 KB
    float*    k_f   = (float*)   (ws + 0x3A0000); // [512,512] 1 MB
    int*      idx_w = (int*)     (ws + 0x4A0000); // [16,10]

    float* outA = (float*)d_out;
    float* outV = outA + (size_t)B_ * TOPK * CLIP * C_;
    float* outI = outV + (size_t)B_ * TOPK * CLIP * C_;

    // 1) clip pooling + tanh -> f16 fusion   (the 268 MB streaming pass)
    tss_clip_pool_tanh<<<ROWS, 256, 0, stream>>>(audio, visual, fus_h);

    // 2) weight / query conversions to f16
    tss_f32_to_f16<<<(C_ * TWO_C + 255) / 256, 256, 0, stream>>>(fc_w, fcw_h, C_ * TWO_C);
    tss_f32_to_f16<<<(C_ * C_ + 255) / 256,   256, 0, stream>>>(wq,   wq_h,  C_ * C_);
    tss_f32_to_f16<<<(C_ * C_ + 255) / 256,   256, 0, stream>>>(wk,   wk_h,  C_ * C_);
    tss_f32_to_f16<<<(B_ * C_ + 255) / 256,   256, 0, stream>>>(qst,  qst_h, B_ * C_);

    // 3) kv = tanh(fusion) @ fc_w^T + fc_b   (M=512, N=512, K=1024)
    {
        int tiles = (ROWS / 16) * (C_ / 16);
        tss_wmma_gemm_nt<<<(tiles + 3) / 4, 128, 0, stream>>>(
            fus_h, fcw_h, fc_b, nullptr, kv_h, ROWS, C_, TWO_C);
    }
    // 4) q = qst @ wq^T + bq                 (M=16, N=512, K=512)
    {
        int tiles = (B_ / 16) * (C_ / 16);
        tss_wmma_gemm_nt<<<(tiles + 3) / 4, 128, 0, stream>>>(
            qst_h, wq_h, bq, q_f, nullptr, B_, C_, C_);
    }
    // 5) k = kv @ wk^T + bk                  (M=512, N=512, K=512)
    {
        int tiles = (ROWS / 16) * (C_ / 16);
        tss_wmma_gemm_nt<<<(tiles + 3) / 4, 128, 0, stream>>>(
            kv_h, wk_h, bk, k_f, nullptr, ROWS, C_, C_);
    }

    // 6) attention + softmax + head-mean + sorted top-10
    tss_attn_topk<<<B_, 128, 0, stream>>>(q_f, k_f, idx_w, outI);

    // 7) gather selected clips (84 MB read + 84 MB write, float4 rows)
    tss_gather_rows<<<B_ * TOPK * CLIP, 256, 0, stream>>>(audio, visual, idx_w, outA, outV);
}
